// Attention_3178275799372
// MI455X (gfx1250) — matrix operand order repaired and checked by reading, not compiled, not run on hardware
//
#include <hip/hip_runtime.h>

// ---------------------------------------------------------------------------
// Multi-head causal attention for gfx1250 (MI455X).
// All 5 GEMM stages via v_wmma_f32_16x16x32_bf16 (bf16 operands, f32 accum).
// K/V tiles staged to LDS with global_load_async_to_lds_b128 (ASYNCcnt),
// double-buffered, shared by all 8 waves of a block. ~103 GFLOP vs ~70MB HBM
// traffic @ 23.3 TB/s -> compute bound: keep the WMMA pipe fed from LDS.
// ---------------------------------------------------------------------------

#define SEQ 4096
#define DM  1024
#define NH  16
#define DK  64

typedef __bf16 bf16;
typedef __attribute__((ext_vector_type(16))) __bf16 v16bf;
typedef __attribute__((ext_vector_type(8)))  float  v8f;

__device__ inline v8f wmma_bf16(v16bf a, v16bf b, v8f c) {
  // (neg_a, A, neg_b, B, c_mod, C, reuse_a, reuse_b)
  return __builtin_amdgcn_wmma_f32_16x16x32_bf16(false, a, false, b, (short)0, c,
                                                 false, false);
}

__device__ inline v8f zero8() {
  v8f z;
#pragma unroll
  for (int i = 0; i < 8; ++i) z[i] = 0.0f;
  return z;
}

// ---- fragment helpers (wave32; A/B use contiguous k order, same for both operands) ----
// A 16x32 bf16: lane holds row m = lane&15; element e -> k = e + 16*(lane>>4)
__device__ inline v16bf load_a_bf16(const bf16* __restrict__ base, int ld,
                                    int m0, int k0, int lane) {
  const bf16* row = base + (size_t)(m0 + (lane & 15)) * ld + k0 + 16 * (lane >> 4);
  v16bf a;
#pragma unroll
  for (int e = 0; e < 8; ++e) a[e] = row[e];
#pragma unroll
  for (int e = 0; e < 8; ++e) a[e + 8] = row[8 + e];
  return a;
}

__device__ inline v16bf load_a_f32(const float* __restrict__ base, int ld,
                                   int m0, int k0, int lane) {
  const float* row = base + (size_t)(m0 + (lane & 15)) * ld + k0 + 16 * (lane >> 4);
  v16bf a;
#pragma unroll
  for (int e = 0; e < 8; ++e) a[e] = (bf16)row[e];
#pragma unroll
  for (int e = 0; e < 8; ++e) a[e + 8] = (bf16)row[8 + e];
  return a;
}

// C/D 16x16 f32: lane holds col n = lane&15; VGPR r -> row m = r + 8*(lane>>4)
__device__ inline void store_c_bf16(bf16* __restrict__ base, int ld, int m0,
                                    int n0, int lane, v8f c) {
  bf16* p = base + (size_t)(m0 + 8 * (lane >> 4)) * ld + n0 + (lane & 15);
#pragma unroll
  for (int r = 0; r < 8; ++r) p[(size_t)r * ld] = (bf16)c[r];
}

__device__ inline void store_c_f32(float* __restrict__ base, int ld, int m0,
                                   int n0, int lane, v8f c) {
  float* p = base + (size_t)(m0 + 8 * (lane >> 4)) * ld + n0 + (lane & 15);
#pragma unroll
  for (int r = 0; r < 8; ++r) p[(size_t)r * ld] = c[r];
}

// ---------------------------------------------------------------------------
// Kernel 0: fp32 -> bf16 conversion (weights, once per launch)
// ---------------------------------------------------------------------------
__global__ void cvt_f32_bf16(const float* __restrict__ s, bf16* __restrict__ d,
                             int n) {
  for (int i = blockIdx.x * blockDim.x + threadIdx.x; i < n;
       i += gridDim.x * blockDim.x)
    d[i] = (bf16)s[i];
}

// ---------------------------------------------------------------------------
// Kernel 1: fused per-head Q/K/V projections. grid=(SEQ/128, H, 3), 256 thr.
// Weight 32x64 k-tile staged TRANSPOSED into LDS (sW[n][k]) so each wave's
// B fragments are contiguous 32B ds reads; shared by all 8 waves.
// ---------------------------------------------------------------------------
__global__ __launch_bounds__(256) void proj_kernel(
    const float* __restrict__ Q, const float* __restrict__ K,
    const float* __restrict__ V, const bf16* __restrict__ WQ,
    const bf16* __restrict__ WK, const bf16* __restrict__ WV,
    bf16* __restrict__ Qh, bf16* __restrict__ Kh, bf16* __restrict__ Vh) {
  __shared__ __align__(16) bf16 sW[64][40];  // [n][k], padded: 80B row stride

  const int tid = threadIdx.x;
  const int lane = tid & 31;
  const int wv = tid >> 5;
  const int m0 = (blockIdx.x * 8 + wv) * 16;
  const int h = blockIdx.y;

  const float* src;
  const bf16* wgt;
  bf16* dst;
  if (blockIdx.z == 0)      { src = Q; wgt = WQ; dst = Qh; }
  else if (blockIdx.z == 1) { src = K; wgt = WK; dst = Kh; }
  else                      { src = V; wgt = WV; dst = Vh; }
  wgt += (size_t)h * DM * DK;
  dst += (size_t)h * SEQ * DK;

  const int kk = tid >> 3;      // 0..31: k row within tile
  const int seg = tid & 7;      // 8-element column segment

  v8f acc[4];
#pragma unroll
  for (int t = 0; t < 4; ++t) acc[t] = zero8();

  for (int k0 = 0; k0 < DM; k0 += 32) {
    __syncthreads();  // previous tile fully consumed
    // stage W[k0+kk][seg*8 .. seg*8+7] -> sW[n][kk] (transposed)
    {
      const bf16* wrow = wgt + (size_t)(k0 + kk) * DK + seg * 8;
      bf16 tmp[8];
#pragma unroll
      for (int e = 0; e < 8; ++e) tmp[e] = wrow[e];
#pragma unroll
      for (int e = 0; e < 8; ++e) sW[seg * 8 + e][kk] = tmp[e];
    }
    __syncthreads();

    v16bf a = load_a_f32(src, DM, m0, k0, lane);
#pragma unroll
    for (int t = 0; t < 4; ++t) {
      // B 32x16: element e -> k = e + 16*(lane>>4), n = t*16 + (lane&15)
      const bf16* p = &sW[t * 16 + (lane & 15)][16 * (lane >> 4)];
      v16bf b;
#pragma unroll
      for (int e = 0; e < 16; ++e) b[e] = p[e];
      acc[t] = wmma_bf16(a, b, acc[t]);
    }
  }
#pragma unroll
  for (int t = 0; t < 4; ++t) store_c_bf16(dst, DK, m0, t * 16, lane, acc[t]);
}

// ---------------------------------------------------------------------------
// Kernel 2: causal flash attention. grid=(SEQ/128, H), 256 thr (8 waves).
// Block owns 128 query rows of one head; K/V 32x64 tiles staged into LDS with
// global_load_async_to_lds_b128 (double-buffered), shared by all waves.
// ---------------------------------------------------------------------------
struct AttnLdsT {
  bf16 K[2][32][72];   // keys x features, padded rows (144B)
  bf16 V[2][32][72];   // keys x features, padded rows
  bf16 P[8][16][40];   // per-wave P staging, padded rows (80B, 16B aligned)
};

__device__ inline void stage_kv(AttnLdsT* lds, const bf16* Kp, const bf16* Vp,
                                int j0, int buf, int tid) {
  // 512 chunks of 16B (K tile 4KB + V tile 4KB); 2 per thread, 2 async
  // instructions per wave. Per-lane: LDS[vdst] = MEM[vaddr] (16B), ASYNCcnt.
#pragma unroll
  for (int c = 0; c < 2; ++c) {
    const int idx = tid + 256 * c;          // 0..511
    const int mat = idx >> 8;               // 0 = K, 1 = V
    const int r = (idx >> 3) & 31;          // key row in tile
    const int ch = idx & 7;                 // 16B chunk in row
    const bf16* g = (mat ? Vp : Kp) + (size_t)(j0 + r) * DK + ch * 8;
    char* lp = mat ? (char*)&lds->V[buf][r][ch * 8]
                   : (char*)&lds->K[buf][r][ch * 8];
    const unsigned loff = (unsigned)(lp - (char*)lds);  // byte offset in LDS
    asm volatile("global_load_async_to_lds_b128 %0, %1, off"
                 :: "v"(loff), "v"(g)
                 : "memory");
  }
}

__global__ __launch_bounds__(256) void attn_kernel(const bf16* __restrict__ Qh,
                                                   const bf16* __restrict__ Kh,
                                                   const bf16* __restrict__ Vh,
                                                   bf16* __restrict__ Cc) {
  __shared__ __align__(16) AttnLdsT lds;

  const int tid = threadIdx.x;
  const int lane = tid & 31;
  const int wv = tid >> 5;
  const int col = lane & 15;
  const int half = lane >> 4;
  const int Q0 = blockIdx.x * 128;
  const int q0 = Q0 + wv * 16;
  const int h = blockIdx.y;

  const bf16* Qp = Qh + (size_t)h * SEQ * DK;
  const bf16* Kp = Kh + (size_t)h * SEQ * DK;
  const bf16* Vp = Vh + (size_t)h * SEQ * DK;

  // per-wave Q tile fragments (16x64), reused for all key tiles
  v16bf aq0 = load_a_bf16(Qp, DK, q0, 0, lane);
  v16bf aq1 = load_a_bf16(Qp, DK, q0, 32, lane);

  v8f o[4];
#pragma unroll
  for (int t = 0; t < 4; ++t) o[t] = zero8();
  float mi[8], li[8];
#pragma unroll
  for (int r = 0; r < 8; ++r) { mi[r] = -3.0e38f; li[r] = 0.0f; }

  const int Jmax = Q0 + 112;  // largest q0 in this block; tile needed iff j0<=q0
  stage_kv(&lds, Kp, Vp, /*j0=*/0, /*buf=*/0, tid);

  for (int j0 = 0; j0 <= Jmax; j0 += 32) {
    const int buf = (j0 >> 5) & 1;
    asm volatile("s_wait_asynccnt 0x0" ::: "memory");  // this tile landed
    __syncthreads();                                   // visible to all waves
    if (j0 + 32 <= Jmax) stage_kv(&lds, Kp, Vp, j0 + 32, buf ^ 1, tid);

    if (q0 >= j0) {  // wave-uniform causal guard
      // ---- S = Q * K^T : two 16x16 tiles (keys j0+0..15, j0+16..31) ----
      v8f slo = zero8(), shi = zero8();
#pragma unroll
      for (int sub = 0; sub < 2; ++sub) {   // n0 = sub*16
        v8f s = zero8();
#pragma unroll
        for (int kf = 0; kf < 2; ++kf) {    // feature k0 = kf*32
          // B^T fragment: element e -> k = kf*32 + 16*half + e (contiguous)
          const bf16* p = &lds.K[buf][sub * 16 + col][kf * 32 + 16 * half];
          v16bf b;
#pragma unroll
          for (int e = 0; e < 16; ++e) b[e] = p[e];
          s = wmma_bf16(kf ? aq1 : aq0, b, s);
        }
        if (sub == 0) slo = s; else shi = s;
      }

      // ---- online softmax (f32), rows reduced across 16 column lanes ----
#pragma unroll
      for (int r = 0; r < 8; ++r) {
        const int qi = q0 + r + 8 * half;
        float vlo = (j0 + col <= qi) ? slo[r] * 0.125f : -1.0e9f;
        float vhi = (j0 + 16 + col <= qi) ? shi[r] * 0.125f : -1.0e9f;

        float t = fmaxf(vlo, vhi);
        t = fmaxf(t, __shfl_xor(t, 1));
        t = fmaxf(t, __shfl_xor(t, 2));
        t = fmaxf(t, __shfl_xor(t, 4));
        t = fmaxf(t, __shfl_xor(t, 8));

        const float nm = fmaxf(mi[r], t);
        const float sc = __expf(mi[r] - nm);
        const float plo = __expf(vlo - nm);
        const float phi = __expf(vhi - nm);
        float rs = plo + phi;
        rs += __shfl_xor(rs, 1);
        rs += __shfl_xor(rs, 2);
        rs += __shfl_xor(rs, 4);
        rs += __shfl_xor(rs, 8);

        li[r] = li[r] * sc + rs;
        mi[r] = nm;
        o[0][r] *= sc;
        o[1][r] *= sc;
        o[2][r] *= sc;
        o[3][r] *= sc;

        // stage P (C layout) row-major into wave-private LDS region
        lds.P[wv][r + 8 * half][col] = (bf16)plo;
        lds.P[wv][r + 8 * half][col + 16] = (bf16)phi;
      }
      asm volatile("s_wait_dscnt 0x0" ::: "memory");

      // reload P as 16x32 A fragment (contiguous, 16B aligned)
      v16bf ap;
      {
        const int kb = 16 * half;
#pragma unroll
        for (int e = 0; e < 8; ++e) ap[e] = lds.P[wv][col][kb + e];
#pragma unroll
        for (int e = 0; e < 8; ++e) ap[e + 8] = lds.P[wv][col][kb + 8 + e];
      }

      // ---- O += P * V  (V tile 32 keys x 64 features, from LDS) ----
#pragma unroll
      for (int t4 = 0; t4 < 4; ++t4) {
        v16bf b;
#pragma unroll
        for (int e = 0; e < 16; ++e)
          b[e] = lds.V[buf][16 * half + e][t4 * 16 + col];
        o[t4] = wmma_bf16(ap, b, o[t4]);
      }
    }
  }

  // normalize; write to concat buffer columns [h*64, h*64+64)
#pragma unroll
  for (int r = 0; r < 8; ++r) {
    const float inv = 1.0f / li[r];
    o[0][r] *= inv;
    o[1][r] *= inv;
    o[2][r] *= inv;
    o[3][r] *= inv;
  }
#pragma unroll
  for (int t4 = 0; t4 < 4; ++t4)
    store_c_bf16(Cc, DM, q0, h * DK + t4 * 16, lane, o[t4]);
}

// ---------------------------------------------------------------------------
// Kernel 3: output projection out = concat @ W_O (f32 result).
// grid=(SEQ/128, DM/64), 256 thr. Same transposed weight-tile staging.
// ---------------------------------------------------------------------------
__global__ __launch_bounds__(256) void out_kernel(const bf16* __restrict__ Cc,
                                                  const bf16* __restrict__ WO,
                                                  float* __restrict__ out) {
  __shared__ __align__(16) bf16 sW[64][40];

  const int tid = threadIdx.x;
  const int lane = tid & 31;
  const int wv = tid >> 5;
  const int m0 = (blockIdx.x * 8 + wv) * 16;
  const int n0 = blockIdx.y * 64;
  const int kk = tid >> 3;
  const int seg = tid & 7;

  v8f acc[4];
#pragma unroll
  for (int t = 0; t < 4; ++t) acc[t] = zero8();

  for (int k0 = 0; k0 < DM; k0 += 32) {
    __syncthreads();
    {
      const bf16* wrow = WO + (size_t)(k0 + kk) * DM + n0 + seg * 8;
      bf16 tmp[8];
#pragma unroll
      for (int e = 0; e < 8; ++e) tmp[e] = wrow[e];
#pragma unroll
      for (int e = 0; e < 8; ++e) sW[seg * 8 + e][kk] = tmp[e];
    }
    __syncthreads();

    v16bf a = load_a_bf16(Cc, DM, m0, k0, lane);
#pragma unroll
    for (int t = 0; t < 4; ++t) {
      const bf16* p = &sW[t * 16 + (lane & 15)][16 * (lane >> 4)];
      v16bf b;
#pragma unroll
      for (int e = 0; e < 16; ++e) b[e] = p[e];
      acc[t] = wmma_bf16(a, b, acc[t]);
    }
  }
#pragma unroll
  for (int t = 0; t < 4; ++t)
    store_c_f32(out, DM, m0, n0 + t * 16, lane, acc[t]);
}

// ---------------------------------------------------------------------------
extern "C" void kernel_launch(void* const* d_in, const int* in_sizes, int n_in,
                              void* d_out, int out_size, void* d_ws,
                              size_t ws_size, hipStream_t stream) {
  const float* Q = (const float*)d_in[0];
  const float* K = (const float*)d_in[1];
  const float* V = (const float*)d_in[2];
  const float* WQ = (const float*)d_in[3];
  const float* WK = (const float*)d_in[4];
  const float* WV = (const float*)d_in[5];
  const float* WO = (const float*)d_in[6];
  // d_in[7] is the python-scalar mask (==1): causal masking implemented.

  // workspace layout (bf16 elements), 40MB total
  const size_t W_ELEMS = (size_t)NH * DM * DK;   // 1,048,576
  const size_t H_ELEMS = (size_t)NH * SEQ * DK;  // 4,194,304
  bf16* WQb = (bf16*)d_ws;
  bf16* WKb = WQb + W_ELEMS;
  bf16* WVb = WKb + W_ELEMS;
  bf16* WOb = WVb + W_ELEMS;              // DM*DM elems
  bf16* Qh = WOb + (size_t)DM * DM;
  bf16* Kh = Qh + H_ELEMS;
  bf16* Vh = Kh + H_ELEMS;
  bf16* Cc = Vh + H_ELEMS;                // SEQ x DM concat buffer

  cvt_f32_bf16<<<512, 256, 0, stream>>>(WQ, WQb, (int)W_ELEMS);
  cvt_f32_bf16<<<512, 256, 0, stream>>>(WK, WKb, (int)W_ELEMS);
  cvt_f32_bf16<<<512, 256, 0, stream>>>(WV, WVb, (int)W_ELEMS);
  cvt_f32_bf16<<<512, 256, 0, stream>>>(WO, WOb, DM * DM);

  proj_kernel<<<dim3(SEQ / 128, NH, 3), 256, 0, stream>>>(Q, K, V, WQb, WKb,
                                                          WVb, Qh, Kh, Vh);
  attn_kernel<<<dim3(SEQ / 128, NH), 256, 0, stream>>>(Qh, Kh, Vh, Cc);
  out_kernel<<<dim3(SEQ / 128, DM / 64), 256, 0, stream>>>(Cc, WOb,
                                                           (float*)d_out);
}
